// SingleToPairwise_22041772163329
// MI455X (gfx1250) — matrix operand order repaired and checked by reading, not compiled, not run on hardware
//
#include <hip/hip_runtime.h>
#include <math.h>

#define HEADS 32
#define DP    128
#define DIM   768
#define NSEQ  1024
#define PPOS  2048
#define QKW   (2*HEADS*DP)   /* 8192 */
#define RELW  (HEADS*DP)     /* 4096 */

typedef __bf16 bf16_t;
typedef bf16_t v16bf __attribute__((ext_vector_type(16)));
typedef float  v8f   __attribute__((ext_vector_type(8)));
typedef unsigned int v4u __attribute__((ext_vector_type(4)));

union V16U { v16bf v; v4u q[2]; };

// ---- fragment loads from pre-split bf16 arrays (pure b128 vmem, no VALU) ----
// A-operand (16x32): k = kb + e, 16 contiguous elems (two 16B chunks)
__device__ __forceinline__ v16bf load_a16(const bf16_t* __restrict__ p, int kb) {
  V16U u;
  u.q[0] = *(const v4u*)(p + kb);
  u.q[1] = *(const v4u*)(p + kb + 8);
  return u.v;
}
// B-operand (32x16): k = kB + e, 16 contiguous elems (32B)
__device__ __forceinline__ v16bf load_b16(const bf16_t* __restrict__ p, int kB) {
  V16U u;
  u.q[0] = *(const v4u*)(p + kB);
  u.q[1] = *(const v4u*)(p + kB + 8);
  return u.v;
}
__device__ __forceinline__ v16bf zero16() {
  V16U u; u.q[0] = v4u{0,0,0,0}; u.q[1] = v4u{0,0,0,0}; return u.v;
}
// bf16x3 compensated product: a*b ~= ah*bh + ah*bl + al*bh (fp32 accumulate)
__device__ __forceinline__ v8f wmma3(v16bf ah, v16bf al, v16bf bh, v16bf bl, v8f c) {
  c = __builtin_amdgcn_wmma_f32_16x16x32_bf16(false, ah, false, bh, (short)0, c, false, false);
  c = __builtin_amdgcn_wmma_f32_16x16x32_bf16(false, ah, false, bl, (short)0, c, false, false);
  c = __builtin_amdgcn_wmma_f32_16x16x32_bf16(false, al, false, bh, (short)0, c, false, false);
  return c;
}
// split fp32 -> (hi, lo) bf16 pair; compiler picks hw cvt ops
__device__ __forceinline__ void split2(float x, bf16_t& h, bf16_t& l) {
  bf16_t hh = (bf16_t)x;
  h = hh;
  l = (bf16_t)(x - (float)hh);
}

// ---------------- generic fp32 -> hi/lo split (optional broadcast row-bias add) ----------------
__global__ __launch_bounds__(256) void split_kernel(
    const float* __restrict__ src, const float* __restrict__ addv, unsigned addmask,
    bf16_t* __restrict__ hi, bf16_t* __restrict__ lo, int n) {
  int idx = blockIdx.x * 256 + threadIdx.x;
  if (idx >= n) return;
  float x = src[idx];
  if (addv) x += addv[idx & addmask];
  bf16_t h, l; split2(x, h, l);
  hi[idx] = h; lo[idx] = l;
}

// ---------------- pooling + exact GELU, outputs pre-split ----------------
__global__ __launch_bounds__(256) void pool_gelu_kernel(
    const float* __restrict__ single,
    bf16_t* __restrict__ ph, bf16_t* __restrict__ pl,
    bf16_t* __restrict__ gh, bf16_t* __restrict__ gl) {
  int idx = blockIdx.x * 256 + threadIdx.x;          // [0, 1024*768)
  int nrow = idx / DIM;
  int d = idx - nrow * DIM;
  const float* p = single + (size_t)nrow * 16 * DIM + d;
  float s = 0.f;
  #pragma unroll
  for (int t = 0; t < 16; ++t) s += p[(size_t)t * DIM];
  s *= (1.0f / 16.0f);
  bf16_t h, l;
  split2(s, h, l); ph[idx] = h; pl[idx] = l;
  float g = 0.5f * s * (1.0f + erff(s * 0.70710678118654752f));  // exact gelu
  split2(g, h, l); gh[idx] = h; gl[idx] = l;
}

// ---------------- generic C = A @ B^T (+bias), bf16x3 WMMA, pre-split operands ----------------
// A: MxK (lda), B: NxK (ldb), C fp32: MxN (ldc). Wave tile 16x64, block = 4 waves.
__global__ __launch_bounds__(128) void gemm_bf16x3(
    const bf16_t* __restrict__ Ahi, const bf16_t* __restrict__ Alo, int lda,
    const bf16_t* __restrict__ Bhi, const bf16_t* __restrict__ Blo, int ldb,
    const float* __restrict__ bias, float* __restrict__ C, int ldc, int K) {
  int lane = threadIdx.x & 31;
  int wave = threadIdx.x >> 5;
  int n0 = blockIdx.x * 64;
  int m0 = blockIdx.y * 64 + wave * 16;
  int m = lane & 15, half = lane >> 4;
  int kbA = half * 16, kbB = half * 16;
  size_t aoff = (size_t)(m0 + m) * lda;
  v8f acc[4] = {};
  for (int kt = 0; kt < K; kt += 32) {
    v16bf ah = load_a16(Ahi + aoff + kt, kbA);
    v16bf al = load_a16(Alo + aoff + kt, kbA);
    #pragma unroll
    for (int t = 0; t < 4; ++t) {
      size_t boff = (size_t)(n0 + t * 16 + m) * ldb + kt;
      v16bf bh = load_b16(Bhi + boff, kbB);
      v16bf bl = load_b16(Blo + boff, kbB);
      acc[t] = wmma3(ah, al, bh, bl, acc[t]);
    }
  }
  #pragma unroll
  for (int t = 0; t < 4; ++t) {
    #pragma unroll
    for (int v = 0; v < 8; ++v) {
      int row = m0 + v + 8 * half;        // C layout: M = v + 8*(lane>>4)
      int col = n0 + t * 16 + m;          // N = lane&15
      float bv = bias ? bias[col] : 0.f;
      C[(size_t)row * ldc + col] = acc[t][v] + bv;
    }
  }
}

// 16x32 rel window: W[r][c] = sum_k Abiased[r][k] * R[jjbase+c][k], OOB R rows -> 0
__device__ __forceinline__ void window16x32(
    const bf16_t* __restrict__ Ahi, const bf16_t* __restrict__ Alo,  // row r0 base (pre-offset)
    const bf16_t* __restrict__ Rhi, const bf16_t* __restrict__ Rlo,  // head col pre-offset
    int jjbase, float* __restrict__ win, int lane) {
  int m = lane & 15, half = lane >> 4;
  int kbA = half * 16, kbB = half * 16;
  size_t aoff = (size_t)m * QKW;
  v8f acc0 = {}, acc1 = {};
  #pragma unroll
  for (int kt = 0; kt < DP; kt += 32) {
    v16bf ah = load_a16(Ahi + aoff + kt, kbA);
    v16bf al = load_a16(Alo + aoff + kt, kbA);
    int jj0 = jjbase + m;
    v16bf bh0, bl0;
    if (jj0 >= 0 && jj0 < PPOS) {
      size_t o = (size_t)jj0 * RELW + kt;
      bh0 = load_b16(Rhi + o, kbB); bl0 = load_b16(Rlo + o, kbB);
    } else { bh0 = zero16(); bl0 = zero16(); }
    acc0 = wmma3(ah, al, bh0, bl0, acc0);
    int jj1 = jjbase + 16 + m;
    v16bf bh1, bl1;
    if (jj1 >= 0 && jj1 < PPOS) {
      size_t o = (size_t)jj1 * RELW + kt;
      bh1 = load_b16(Rhi + o, kbB); bl1 = load_b16(Rlo + o, kbB);
    } else { bh1 = zero16(); bl1 = zero16(); }
    acc1 = wmma3(ah, al, bh1, bl1, acc1);
  }
  #pragma unroll
  for (int v = 0; v < 8; ++v) {
    int row = v + 8 * half;
    win[row * 33 + m]      = acc0[v];
    win[row * 33 + 16 + m] = acc1[v];
  }
}

// ---------------- fused sim: qk dot + shifted/cropped rel terms, split output ----------------
// simval[i][j][h] = q_h[i].k_h[j] + 0.5*(RQshift[i][j] + RKshift[j][i])
// sim2 layout: per (i,j): 32 x hi | 32 x lo  (128B)
__global__ __launch_bounds__(128) void sim_kernel(
    const bf16_t* __restrict__ qkh,  const bf16_t* __restrict__ qkl,   // 1024 x 8192 split
    const bf16_t* __restrict__ qkbh, const bf16_t* __restrict__ qkbl,  // qk + rel-pos-bias, split
    const bf16_t* __restrict__ reh,  const bf16_t* __restrict__ rel_,  // 2048 x 4096 split
    bf16_t* __restrict__ sim2) {
  __shared__ float smem[4][4][16 * 33];
  int lane = threadIdx.x & 31;
  int ty = threadIdx.x >> 5;               // wave in block = head sub-index
  int h = blockIdx.z * 4 + ty;
  int j0 = blockIdx.x * 16;
  int i0 = blockIdx.y * 16;
  int m = lane & 15, half = lane >> 4;
  int kbA = half * 16, kbB = half * 16;
  int qc = h * DP;                  // q column base
  int kc = HEADS * DP + h * DP;     // k column base

  // q.k part over K=128
  v8f accS = {};
  size_t aoff = (size_t)(i0 + m) * QKW + qc;
  size_t boff = (size_t)(j0 + m) * QKW + kc;
  #pragma unroll
  for (int kt = 0; kt < DP; kt += 32) {
    v16bf ah = load_a16(qkh + aoff + kt, kbA);
    v16bf al = load_a16(qkl + aoff + kt, kbA);
    v16bf bh = load_b16(qkh + boff + kt, kbB);
    v16bf bl = load_b16(qkl + boff + kt, kbB);
    accS = wmma3(ah, al, bh, bl, accS);
  }

  // rel windows. delta = j - i. Shift map (relative_shift + crop):
  //   delta<=512 : REL[i][1535+delta] ; delta==513 : 0 ; delta>=514 : REL[i+1][delta-514]
  int dbase = j0 - i0;
  float* wQn = smem[ty][0];
  float* wQw = smem[ty][1];
  float* wKn = smem[ty][2];
  float* wKw = smem[ty][3];
  const bf16_t* Rh = reh + h * DP;
  const bf16_t* Rl = rel_ + h * DP;
  if (dbase <= 527)
    window16x32(qkbh + (size_t)i0 * QKW + qc,       qkbl + (size_t)i0 * QKW + qc,       Rh, Rl, 1520 + dbase, wQn, lane);
  if (dbase >= 499)
    window16x32(qkbh + (size_t)(i0 + 1) * QKW + qc, qkbl + (size_t)(i0 + 1) * QKW + qc, Rh, Rl, dbase - 529,  wQw, lane);
  if (dbase >= -527)
    window16x32(qkbh + (size_t)j0 * QKW + kc,       qkbl + (size_t)j0 * QKW + kc,       Rh, Rl, 1520 - dbase, wKn, lane);
  if (dbase <= -499)
    window16x32(qkbh + (size_t)(j0 + 1) * QKW + kc, qkbl + (size_t)(j0 + 1) * QKW + kc, Rh, Rl, -dbase - 529, wKw, lane);
  __syncthreads();

  #pragma unroll
  for (int v = 0; v < 8; ++v) {
    int mloc = v + 8 * half;
    int i = i0 + mloc, j = j0 + m;
    int delta = j - i;
    float rq, rk;
    if (delta <= 512)      rq = wQn[mloc * 33 + (m - mloc + 15)];
    else if (delta == 513) rq = 0.f;
    else                   rq = wQw[mloc * 33 + (m - mloc + 15)];
    int d2 = -delta;
    if (d2 <= 512)         rk = wKn[m * 33 + (mloc - m + 15)];
    else if (d2 == 513)    rk = 0.f;
    else                   rk = wKw[m * 33 + (mloc - m + 15)];
    float val = accS[v] + 0.5f * (rq + rk);
    bf16_t hh, ll; split2(val, hh, ll);
    size_t base = ((size_t)i * NSEQ + j) * 64;
    sim2[base + h] = hh;
    sim2[base + 32 + h] = ll;
  }
}

// ---------------- final: out = sim @ w_pair^T + b_pair + outer_q[i] + outer_k[j] ----------------
__global__ __launch_bounds__(256) void final_kernel(
    const bf16_t* __restrict__ sim2,   // (1024,1024, 32hi|32lo)
    const bf16_t* __restrict__ wph, const bf16_t* __restrict__ wpl,  // (128,32) split
    const float* __restrict__ bpair,   // (128)
    const float* __restrict__ outer,   // (1024,256): [:,0:128]=q, [:,128:256]=k
    float* __restrict__ out) {         // (1024,1024,128)
  int lane = threadIdx.x & 31;
  int wv = threadIdx.x >> 5;
  int wid = blockIdx.x * 8 + wv;      // 65536 strips of 16 j x 128 d
  int i = wid >> 6;
  int j0 = (wid & 63) * 16;
  int m = lane & 15, half = lane >> 4;
  int kbA = half * 16, kbB = half * 16;
  const bf16_t* simrow = sim2 + ((size_t)i * NSEQ + j0 + m) * 64;
  v16bf ah = load_a16(simrow, kbA);        // K = 32 heads
  v16bf al = load_a16(simrow + 32, kbA);
  #pragma unroll
  for (int dt = 0; dt < 8; ++dt) {
    int d = dt * 16 + m;
    v16bf bh = load_b16(wph + (size_t)d * HEADS, kbB);
    v16bf bl = load_b16(wpl + (size_t)d * HEADS, kbB);
    v8f c = {};
    c = wmma3(ah, al, bh, bl, c);
    float addi = outer[(size_t)i * 256 + d] + bpair[d];
    #pragma unroll
    for (int v = 0; v < 8; ++v) {
      int j = j0 + v + 8 * half;
      out[((size_t)i * NSEQ + j) * DP + d] = c[v] + addi + outer[(size_t)j * 256 + 128 + d];
    }
  }
}

extern "C" void kernel_launch(void* const* d_in, const int* in_sizes, int n_in,
                              void* d_out, int out_size, void* d_ws, size_t ws_size,
                              hipStream_t stream) {
  (void)in_sizes; (void)n_in; (void)out_size;
  const float* single = (const float*)d_in[0];
  const float* relpos = (const float*)d_in[1];
  const float* wqk    = (const float*)d_in[2];
  const float* wout   = (const float*)d_in[3];
  const float* wpair  = (const float*)d_in[4];
  const float* bpair  = (const float*)d_in[5];
  const float* wrel   = (const float*)d_in[6];
  const float* brel   = (const float*)d_in[7];
  const float* qkbias = (const float*)d_in[8];

  char* w = (char*)d_ws;
  size_t used = 0;
  auto alloc = [&](size_t bytes) {
    char* p = w + used;
    used += (bytes + 255) & ~(size_t)255;
    return p;
  };
  const size_t NPOOL = (size_t)1024 * 768;
  const size_t NRP   = (size_t)2048 * 768;
  const size_t NWQK  = (size_t)8192 * 768;
  const size_t NWREL = (size_t)4096 * 768;
  const size_t NWOUT = (size_t)256 * 768;
  const size_t NQK   = (size_t)1024 * 8192;
  const size_t NRE   = (size_t)2048 * 4096;

  bf16_t* ph    = (bf16_t*)alloc(NPOOL * 2);
  bf16_t* pl    = (bf16_t*)alloc(NPOOL * 2);
  bf16_t* gh    = (bf16_t*)alloc(NPOOL * 2);
  bf16_t* gl    = (bf16_t*)alloc(NPOOL * 2);
  bf16_t* rph   = (bf16_t*)alloc(NRP * 2);
  bf16_t* rpl   = (bf16_t*)alloc(NRP * 2);
  bf16_t* wqkh  = (bf16_t*)alloc(NWQK * 2);
  bf16_t* wqkl  = (bf16_t*)alloc(NWQK * 2);
  bf16_t* wrelh = (bf16_t*)alloc(NWREL * 2);
  bf16_t* wrell = (bf16_t*)alloc(NWREL * 2);
  bf16_t* wouth = (bf16_t*)alloc(NWOUT * 2);
  bf16_t* woutl = (bf16_t*)alloc(NWOUT * 2);
  bf16_t* wph   = (bf16_t*)alloc(4096 * 2);
  bf16_t* wpl   = (bf16_t*)alloc(4096 * 2);
  float*  stage = (float*)alloc(NQK * 4);            // reused: qk fp32, then relenc fp32
  bf16_t* qkh   = (bf16_t*)alloc(NQK * 2);
  bf16_t* qkl   = (bf16_t*)alloc(NQK * 2);
  bf16_t* qkbh  = (bf16_t*)alloc(NQK * 2);
  bf16_t* qkbl  = (bf16_t*)alloc(NQK * 2);
  bf16_t* reh   = (bf16_t*)alloc(NRE * 2);
  bf16_t* rel_  = (bf16_t*)alloc(NRE * 2);
  float*  outerb= (float*)alloc((size_t)1024 * 256 * 4);
  bf16_t* sim2  = (bf16_t*)alloc((size_t)1024 * 1024 * 64 * 2);
  if (ws_size < used) return;                         // ~306 MB scratch required
  float* out = (float*)d_out;

  // inputs -> split bf16
  pool_gelu_kernel<<<3072, 256, 0, stream>>>(single, ph, pl, gh, gl);
  split_kernel<<<(int)(NRP / 256), 256, 0, stream>>>(relpos, nullptr, 0, rph, rpl, (int)NRP);
  split_kernel<<<(int)(NWQK / 256), 256, 0, stream>>>(wqk, nullptr, 0, wqkh, wqkl, (int)NWQK);
  split_kernel<<<(int)(NWREL / 256), 256, 0, stream>>>(wrel, nullptr, 0, wrelh, wrell, (int)NWREL);
  split_kernel<<<(int)(NWOUT / 256), 256, 0, stream>>>(wout, nullptr, 0, wouth, woutl, (int)NWOUT);
  split_kernel<<<16, 256, 0, stream>>>(wpair, nullptr, 0, wph, wpl, 4096);

  // qk = pooled @ w_qk^T ; split plain and (+rel-pos-bias) variants
  gemm_bf16x3<<<dim3(128, 16), 128, 0, stream>>>(ph, pl, DIM, wqkh, wqkl, DIM, nullptr, stage, QKW, DIM);
  split_kernel<<<(int)(NQK / 256), 256, 0, stream>>>(stage, nullptr, 0, qkh, qkl, (int)NQK);
  split_kernel<<<(int)(NQK / 256), 256, 0, stream>>>(stage, qkbias, QKW - 1, qkbh, qkbl, (int)NQK);

  // rel_enc = rel_pos_feats @ w_rel^T + b_rel ; split
  gemm_bf16x3<<<dim3(64, 32), 128, 0, stream>>>(rph, rpl, DIM, wrelh, wrell, DIM, brel, stage, RELW, DIM);
  split_kernel<<<(int)(NRE / 256), 256, 0, stream>>>(stage, nullptr, 0, reh, rel_, (int)NRE);

  // outer = gelu(pooled) @ w_outer^T (fp32, consumed as scalars)
  gemm_bf16x3<<<dim3(4, 16), 128, 0, stream>>>(gh, gl, DIM, wouth, woutl, DIM, nullptr, outerb, 256, DIM);

  // fused sim + relative shift ; then WMMA pair projection + outer-sum epilogue
  sim_kernel<<<dim3(64, 64, 8), 128, 0, stream>>>(qkh, qkl, qkbh, qkbl, reh, rel_, sim2);
  final_kernel<<<8192, 256, 0, stream>>>(sim2, wph, wpl, bpair, outerb, out);
}
